// RayMarcher_46901042873049
// MI455X (gfx1250) — matrix operand order repaired and checked by reading, not compiled, not run on hardware
//
#include <hip/hip_runtime.h>

typedef __attribute__((ext_vector_type(16))) _Float16 v16h;
typedef __attribute__((ext_vector_type(8)))  _Float16 v8h;
typedef __attribute__((ext_vector_type(2)))  _Float16 h2;
typedef __attribute__((ext_vector_type(8)))  float    v8f;
typedef _Float16 H;

#define WMMA(a, b, c) __builtin_amdgcn_wmma_f32_16x16x32_f16(false, (a), false, (b), (short)0, (c), false, false)

// Internal enc layout (cols 0..63): [6*band + 2*axis] = sin, +1 = cos (bands 0..9),
// cols 60..62 = raw pos xyz, col 63 = zero pad.
// Reference layout: [x,y,z, then per band: sin_x,sin_y,sin_z,cos_x,cos_y,cos_z].
__device__ __forceinline__ int enc_ref_k(int c) {     // internal col -> reference row of W1
    if (c >= 60) return c - 60;                        // pos xyz
    int band = c / 6, rem = c % 6, axis = rem >> 1;
    return 3 + 6 * band + ((rem & 1) ? 3 : 0) + axis;
}

// Internal rgb_in layout: cols 0..23 = vd sin/cos pairs (4 bands), 24..26 = vd xyz,
// 27..41 = feat 0..14, 42..63 = zero pad.
// Reference rgb_in: [feat(15), vd_x,vd_y,vd_z, per band sin_xyz cos_xyz].
__device__ __forceinline__ int rgb_ref_k(int c) {     // internal col -> reference row of V1
    if (c < 24) { int band = c / 6, rem = c % 6, axis = rem >> 1;
                  return 15 + 3 + 6 * band + ((rem & 1) ? 3 : 0) + axis; }
    if (c < 27) return 15 + (c - 24);                  // vd xyz
    return c - 27;                                     // features
}

// Load A operand (16x64 tile in LDS, row stride 64 halfs) as two 16x16x32 A chunks.
__device__ __forceinline__ void load_a64(const H* row_base, int hi, v16h& a0, v16h& a1) {
    v8h* p0 = (v8h*)&a0;
    p0[0] = *(const v8h*)(row_base + hi * 16);
    p0[1] = *(const v8h*)(row_base + 8 + hi * 16);
    v8h* p1 = (v8h*)&a1;
    p1[0] = *(const v8h*)(row_base + 32 + hi * 16);
    p1[1] = *(const v8h*)(row_base + 40 + hi * 16);
}

__global__ __launch_bounds__(256) void dt_kernel(const float* __restrict__ tmin,
                                                 const float* __restrict__ tmax,
                                                 float* __restrict__ ws, int n) {
    __shared__ float red[256];
    float s = 0.0f;
    for (int i = threadIdx.x; i < n; i += 256) s += tmax[i] - tmin[i];
    red[threadIdx.x] = s;
    __syncthreads();
    for (int o = 128; o > 0; o >>= 1) {
        if ((int)threadIdx.x < o) red[threadIdx.x] += red[threadIdx.x + o];
        __syncthreads();
    }
    if (threadIdx.x == 0) ws[0] = red[0] / (float)n * (1.0f / 128.0f);
}

__global__ __launch_bounds__(128)
void nerf_kernel(const float* __restrict__ orig, const float* __restrict__ dirs,
                 const float* __restrict__ tmin, const float* __restrict__ tmax,
                 const float* __restrict__ W1, const float* __restrict__ b1,
                 const float* __restrict__ W2, const float* __restrict__ b2,
                 const float* __restrict__ V1, const float* __restrict__ c1,
                 const float* __restrict__ V2, const float* __restrict__ c2,
                 const float* __restrict__ dtp, float* __restrict__ out, int N) {
    // Transposed (output-major) f16 weights, K permuted to the internal enc layouts.
    __shared__ __align__(32) H W1t[64][64];
    __shared__ __align__(32) H W2t[16][64];
    __shared__ __align__(32) H V1t[64][64];
    __shared__ __align__(32) H V2t[16][64];
    __shared__ float b1s[64], b2s[16], c1s[64], c2s[4];
    __shared__ __align__(32) H bufEnc[4][16][64];
    __shared__ __align__(32) H bufH[4][16][64];
    __shared__ __align__(32) H bufRGB[4][16][64];
    __shared__ float sigS[4][128];
    __shared__ float rgbS[4][128][3];

    const int tid  = threadIdx.x;
    const int wave = tid >> 5;
    const int lane = tid & 31;
    const int lm   = lane & 15;
    const int hi   = lane >> 4;

    // ---- stage weights (transpose + K permutation, f32->f16, zero-pad) ----
    for (int i = tid; i < 64 * 64; i += 128) {
        int n = i >> 6, c = i & 63;
        W1t[n][c] = (c < 63) ? (H)W1[enc_ref_k(c) * 64 + n] : (H)0.0f;
        V1t[n][c] = (c < 42) ? (H)V1[rgb_ref_k(c) * 64 + n] : (H)0.0f;
    }
    for (int i = tid; i < 16 * 64; i += 128) {
        int n = i >> 6, k = i & 63;
        W2t[n][k] = (H)W2[k * 16 + n];
        V2t[n][k] = (n < 3) ? (H)V2[k * 3 + n] : (H)0.0f;
    }
    if (tid < 64) { b1s[tid] = b1[tid]; c1s[tid] = c1[tid]; }
    if (tid < 16) b2s[tid] = b2[tid];
    if (tid < 4)  c2s[tid] = (tid < 3) ? c2[tid] : 0.0f;

    const int ray = blockIdx.x * 4 + wave;
    const float ox = orig[ray * 3 + 0], oy = orig[ray * 3 + 1], oz = orig[ray * 3 + 2];
    const float dx = dirs[ray * 3 + 0], dy = dirs[ray * 3 + 1], dz = dirs[ray * 3 + 2];
    const float t0 = tmin[ray], t1 = tmax[ray];
    const float dt = dtp[0];

    // ---- prefill view-dir part of rgb_in (constant per ray) ----
    {
        float inv = 1.0f / (sqrtf(dx * dx + dy * dy + dz * dz) + 1e-8f);
        float vx = dx * inv, vy = dy * inv, vz = dz * inv;
        for (int i = lane; i < 16 * 12; i += 32) {          // sin/cos pairs, 4 bands x 3 axes
            int row = i / 12, p = i - row * 12;
            int band = p / 3, axis = p - band * 3;
            float v = (axis == 0) ? vx : ((axis == 1) ? vy : vz);
            float a = (float)(1 << band) * v;
            *(h2*)&bufRGB[wave][row][6 * band + 2 * axis] = (h2){(H)__sinf(a), (H)__cosf(a)};
        }
        for (int i = lane; i < 16 * 3; i += 32) {           // vd xyz at cols 24..26
            int row = i / 3, c = i - row * 3;
            float v = (c == 0) ? vx : ((c == 1) ? vy : vz);
            bufRGB[wave][row][24 + c] = (H)v;
        }
        for (int i = lane; i < 16 * 11; i += 32) {          // zero pad cols 42..63
            int row = i / 11, c = i - row * 11;
            *(h2*)&bufRGB[wave][row][42 + 2 * c] = (h2){(H)0.0f, (H)0.0f};
        }
    }
    __syncthreads();

    for (int s0 = 0; s0 < 128; s0 += 16) {
        // ---- positional encoding: 2 lanes per sample row, 5 bands each, branch-free ----
        {
            const int srow = lm;
            float u = (float)(s0 + srow) * (1.0f / 127.0f);
            float t = t0 + u * (t1 - t0);
            float px = ox + dx * t, py = oy + dy * t, pz = oz + dz * t;
            H* row = &bufEnc[wave][srow][0];
            const float sc = hi ? 32.0f : 1.0f;             // 2^5 offset for bands 5..9
            float fx = px * sc, fy = py * sc, fz = pz * sc;
            const int cb = hi * 30;                         // col base: 6 * (hi*5)
            #pragma unroll
            for (int b = 0; b < 5; ++b) {
                float f = (float)(1 << b);
                float ax = f * fx, ay = f * fy, az = f * fz;
                *(h2*)&row[cb + 6 * b + 0] = (h2){(H)__sinf(ax), (H)__cosf(ax)};
                *(h2*)&row[cb + 6 * b + 2] = (h2){(H)__sinf(ay), (H)__cosf(ay)};
                *(h2*)&row[cb + 6 * b + 4] = (h2){(H)__sinf(az), (H)__cosf(az)};
            }
            if (hi == 0) *(h2*)&row[60] = (h2){(H)px, (H)py};
            else         *(h2*)&row[62] = (h2){(H)pz, (H)0.0f};
        }

        // ---- layer 1: enc(16x64) @ W1(64x64) ----
        v16h a0, a1;
        load_a64(&bufEnc[wave][lm][0], hi, a0, a1);
        v8f acc0 = {}, acc1 = {}, acc2 = {}, acc3 = {};
        acc0 = WMMA(a0, (*(const v16h*)&W1t[ 0 + lm][hi * 16]),      acc0);
        acc0 = WMMA(a1, (*(const v16h*)&W1t[ 0 + lm][32 + hi * 16]), acc0);
        acc1 = WMMA(a0, (*(const v16h*)&W1t[16 + lm][hi * 16]),      acc1);
        acc1 = WMMA(a1, (*(const v16h*)&W1t[16 + lm][32 + hi * 16]), acc1);
        acc2 = WMMA(a0, (*(const v16h*)&W1t[32 + lm][hi * 16]),      acc2);
        acc2 = WMMA(a1, (*(const v16h*)&W1t[32 + lm][32 + hi * 16]), acc2);
        acc3 = WMMA(a0, (*(const v16h*)&W1t[48 + lm][hi * 16]),      acc3);
        acc3 = WMMA(a1, (*(const v16h*)&W1t[48 + lm][32 + hi * 16]), acc3);
        #pragma unroll
        for (int g = 0; g < 8; ++g) {
            int M = g + hi * 8;
            bufH[wave][M][ 0 + lm] = (H)fmaxf(acc0[g] + b1s[ 0 + lm], 0.0f);
            bufH[wave][M][16 + lm] = (H)fmaxf(acc1[g] + b1s[16 + lm], 0.0f);
            bufH[wave][M][32 + lm] = (H)fmaxf(acc2[g] + b1s[32 + lm], 0.0f);
            bufH[wave][M][48 + lm] = (H)fmaxf(acc3[g] + b1s[48 + lm], 0.0f);
        }

        // ---- layer 2: h(16x64) @ W2(64x16) -> sigma | 15 feat ----
        load_a64(&bufH[wave][lm][0], hi, a0, a1);
        v8f o2 = {};
        o2 = WMMA(a0, (*(const v16h*)&W2t[lm][hi * 16]),      o2);
        o2 = WMMA(a1, (*(const v16h*)&W2t[lm][32 + hi * 16]), o2);
        #pragma unroll
        for (int g = 0; g < 8; ++g) {
            int M = g + hi * 8;
            float v = o2[g] + b2s[lm];
            if (lm == 0) sigS[wave][s0 + M] = fmaxf(v, 0.0f);
            else         bufRGB[wave][M][27 + lm - 1] = (H)v;   // feat at cols 27..41
        }

        // ---- layer 3: rgb_in(16x64) @ V1(64x64) ----
        load_a64(&bufRGB[wave][lm][0], hi, a0, a1);
        v8f g0 = {}, g1 = {}, g2 = {}, g3 = {};
        g0 = WMMA(a0, (*(const v16h*)&V1t[ 0 + lm][hi * 16]),      g0);
        g0 = WMMA(a1, (*(const v16h*)&V1t[ 0 + lm][32 + hi * 16]), g0);
        g1 = WMMA(a0, (*(const v16h*)&V1t[16 + lm][hi * 16]),      g1);
        g1 = WMMA(a1, (*(const v16h*)&V1t[16 + lm][32 + hi * 16]), g1);
        g2 = WMMA(a0, (*(const v16h*)&V1t[32 + lm][hi * 16]),      g2);
        g2 = WMMA(a1, (*(const v16h*)&V1t[32 + lm][32 + hi * 16]), g2);
        g3 = WMMA(a0, (*(const v16h*)&V1t[48 + lm][hi * 16]),      g3);
        g3 = WMMA(a1, (*(const v16h*)&V1t[48 + lm][32 + hi * 16]), g3);
        #pragma unroll
        for (int g = 0; g < 8; ++g) {
            int M = g + hi * 8;
            bufH[wave][M][ 0 + lm] = (H)fmaxf(g0[g] + c1s[ 0 + lm], 0.0f);
            bufH[wave][M][16 + lm] = (H)fmaxf(g1[g] + c1s[16 + lm], 0.0f);
            bufH[wave][M][32 + lm] = (H)fmaxf(g2[g] + c1s[32 + lm], 0.0f);
            bufH[wave][M][48 + lm] = (H)fmaxf(g3[g] + c1s[48 + lm], 0.0f);
        }

        // ---- layer 4: h2(16x64) @ V2(64x3 padded) -> sigmoid rgb ----
        load_a64(&bufH[wave][lm][0], hi, a0, a1);
        v8f o4 = {};
        o4 = WMMA(a0, (*(const v16h*)&V2t[lm][hi * 16]),      o4);
        o4 = WMMA(a1, (*(const v16h*)&V2t[lm][32 + hi * 16]), o4);
        if (lm < 3) {
            #pragma unroll
            for (int g = 0; g < 8; ++g) {
                int M = g + hi * 8;
                float v = o4[g] + c2s[lm];
                rgbS[wave][s0 + M][lm] = 1.0f / (1.0f + __expf(-v));
            }
        }
    }

    __syncthreads();

    // ---- per-ray transmittance scan ----
    if (tid < 4) {
        int r = blockIdx.x * 4 + tid;
        float Texcl = 1.0f, Tout = 1.0f;
        float r0 = 0.0f, r1 = 0.0f, r2 = 0.0f;
        for (int s = 0; s < 128; ++s) {
            float alpha = 1.0f - __expf(-sigS[tid][s] * dt);
            float om = 1.0f - alpha;
            bool act = Texcl > 1e-4f;
            float w = act ? Texcl * alpha : 0.0f;
            r0 += w * rgbS[tid][s][0];
            r1 += w * rgbS[tid][s][1];
            r2 += w * rgbS[tid][s][2];
            if (act) Tout *= om;
            Texcl *= om;
        }
        out[r * 3 + 0] = r0;
        out[r * 3 + 1] = r1;
        out[r * 3 + 2] = r2;
        out[N * 3 + r] = Tout;
    }
}

extern "C" void kernel_launch(void* const* d_in, const int* in_sizes, int n_in,
                              void* d_out, int out_size, void* d_ws, size_t ws_size,
                              hipStream_t stream) {
    const float* orig = (const float*)d_in[0];
    const float* dirs = (const float*)d_in[1];
    const float* tmin = (const float*)d_in[2];
    const float* tmax = (const float*)d_in[3];
    const float* W1   = (const float*)d_in[4];
    const float* b1   = (const float*)d_in[5];
    const float* W2   = (const float*)d_in[6];
    const float* b2   = (const float*)d_in[7];
    const float* V1   = (const float*)d_in[8];
    const float* c1   = (const float*)d_in[9];
    const float* V2   = (const float*)d_in[10];
    const float* c2   = (const float*)d_in[11];
    const int N = in_sizes[2];

    float* dt = (float*)d_ws;
    dt_kernel<<<1, 256, 0, stream>>>(tmin, tmax, dt, N);
    nerf_kernel<<<N / 4, 128, 0, stream>>>(orig, dirs, tmin, tmax,
                                           W1, b1, W2, b2, V1, c1, V2, c2,
                                           dt, (float*)d_out, N);
}
